// WindowCrossAttention_8177617731925
// MI455X (gfx1250) — matrix operand order repaired and checked by reading, not compiled, not run on hardware
//
#include <hip/hip_runtime.h>

typedef _Float16 v16h __attribute__((ext_vector_type(16)));
typedef _Float16 v8h  __attribute__((ext_vector_type(8)));
typedef float    v8f  __attribute__((ext_vector_type(8)));

#define WS_Q_OFF   (0UL)
#define WS_KV_OFF  (154140672UL)                         // 200704*384*2
#define WS_X_OFF   (462422016UL)                         // + 200704*768*2
#define WS_WTQ_OFF (616562688UL)                         // + 200704*384*2
#define WS_WTKV_OFF (616857600UL)                        // + 384*384*2
#define WS_WTP_OFF (617447424UL)                         // + 384*768*2

// ---- WMMA fragment loaders (CDNA5 16-bit layouts, wave32) -------------------
// A 16x32 (MxK): lane<16 -> M=lane, K={0..7} then {16..23}; lane>=16 -> K={8..15},{24..31}
__device__ __forceinline__ v16h frag_a_lds(const _Float16* base, int stride, int lane) {
    const int m  = lane & 15;
    const int kb = (lane >> 4) << 3;
    const _Float16* p = base + m * stride + kb;
    union { v16h v; v8h h[2]; } u;
    u.h[0] = *(const v8h*)(p);
    u.h[1] = *(const v8h*)(p + 16);
    return u.v;
}
// B 32x16 (KxN) from LDS stored [n][k]: lane<16 -> N=lane, K={0..7},{16..23}; lane>=16 -> K={8..15},{24..31}
__device__ __forceinline__ v16h frag_b_lds(const _Float16* base, int stride, int lane) {
    const int n  = lane & 15;
    const int kb = (lane >> 4) << 3;
    const _Float16* p = base + n * stride + kb;
    union { v16h v; v8h h[2]; } u;
    u.h[0] = *(const v8h*)(p);
    u.h[1] = *(const v8h*)(p + 16);
    return u.v;
}
__device__ __forceinline__ v8f wmma16(v16h a, v16h b, v8f c) {
    return __builtin_amdgcn_wmma_f32_16x16x32_f16(false, a, false, b, (short)0, c,
                                                  false, false);
}

// ---- one-time weight prep: W[384,N] f32 -> Wt[N,384] f16 --------------------
__global__ __launch_bounds__(256) void transpose_w_kernel(
    const float* __restrict__ W, _Float16* __restrict__ Wt, int Ncols)
{
    const int idx = blockIdx.x * 256 + threadIdx.x;    // over 384*Ncols (exact)
    const int k = idx / Ncols;
    const int n = idx - k * Ncols;
    Wt[n * 384 + k] = (_Float16)W[idx];
}

// ---- tile stage: A[M,384] (f32 or f16) + Wt16[N,384] f16 into LDS -----------
template <typename AT>
__device__ __forceinline__ void load_tiles(
    const AT* __restrict__ A, const _Float16* __restrict__ Wt16,
    long row0, int col0, int k0, _Float16* As, _Float16* Ws, int tid)
{
    // A tile 64x32 halves: thread -> 8 contiguous elements of one row
    {
        const int r = tid >> 2;
        const int c = (tid & 3) * 8;
        const AT* src = A + (row0 + r) * 384 + k0 + c;
        v8h h;
        if constexpr (sizeof(AT) == 4) {
            float4 f0 = *(const float4*)(src);
            float4 f1 = *(const float4*)(src + 4);
            h[0]=(_Float16)f0.x; h[1]=(_Float16)f0.y; h[2]=(_Float16)f0.z; h[3]=(_Float16)f0.w;
            h[4]=(_Float16)f1.x; h[5]=(_Float16)f1.y; h[6]=(_Float16)f1.z; h[7]=(_Float16)f1.w;
        } else {
            h = *(const v8h*)src;
        }
        *(v8h*)&As[r * 40 + c] = h;
    }
    // W tile 128x32 halves from pre-transposed f16: thread -> 16 contiguous K
    {
        const int n = tid >> 1;                        // 0..127
        const int c = (tid & 1) * 16;                  // 0 or 16
        const _Float16* src = Wt16 + (long)(col0 + n) * 384 + k0 + c;
        *(v8h*)&Ws[n * 40 + c]     = *(const v8h*)(src);
        *(v8h*)&Ws[n * 40 + c + 8] = *(const v8h*)(src + 8);
    }
}

// ---- GEMM: C[M,Ncols] = A[M,384] * W + bias, LDS double-buffered ------------
// grid = (Ncols/128, M/64): n-chunk varies fastest so A row-blocks stay in L2.
template <typename AT, typename OT>
__global__ __launch_bounds__(256) void gemm_bias_kernel(
    const AT* __restrict__ A, const _Float16* __restrict__ Wt16,
    const float* __restrict__ bias, OT* __restrict__ C,
    int Ncols, float scale)
{
    __shared__ _Float16 As[2][64 * 40];
    __shared__ _Float16 Ws[2][128 * 40];

    const int tid  = threadIdx.x;
    const int lane = tid & 31;
    const int wid  = tid >> 5;
    const int wm   = wid & 1;
    const int wn   = wid >> 1;
    const int  col0 = blockIdx.x * 128;
    const long row0 = (long)blockIdx.y * 64;

    v8f acc[2][2] = {};

    load_tiles<AT>(A, Wt16, row0, col0, 0, As[0], Ws[0], tid);

    for (int it = 0; it < 12; ++it) {
        __syncthreads();
        if (it < 11)
            load_tiles<AT>(A, Wt16, row0, col0, (it + 1) * 32,
                           As[(it + 1) & 1], Ws[(it + 1) & 1], tid);
        const _Float16* Ab = As[it & 1];
        const _Float16* Wb = Ws[it & 1];
        v16h af[2], bf[2];
        #pragma unroll
        for (int mi = 0; mi < 2; ++mi)
            af[mi] = frag_a_lds(&Ab[(wm * 32 + mi * 16) * 40], 40, lane);
        #pragma unroll
        for (int ni = 0; ni < 2; ++ni)
            bf[ni] = frag_b_lds(&Wb[(wn * 32 + ni * 16) * 40], 40, lane);
        #pragma unroll
        for (int mi = 0; mi < 2; ++mi)
            #pragma unroll
            for (int ni = 0; ni < 2; ++ni)
                acc[mi][ni] = wmma16(af[mi], bf[ni], acc[mi][ni]);
    }

    // Epilogue: C/D layout -> lane<16: M=r, N=lane; lane>=16: M=8+r, N=lane-16
    const int mlane = (lane >> 4) << 3;
    const int nlane = lane & 15;
    float bv[2];
    #pragma unroll
    for (int ni = 0; ni < 2; ++ni)
        bv[ni] = bias[col0 + wn * 32 + ni * 16 + nlane];
    #pragma unroll
    for (int mi = 0; mi < 2; ++mi)
        #pragma unroll
        for (int ni = 0; ni < 2; ++ni)
            #pragma unroll
            for (int r = 0; r < 8; ++r) {
                const long row = row0 + wm * 32 + mi * 16 + mlane + r;
                const int  col = col0 + wn * 32 + ni * 16 + nlane;
                C[row * Ncols + col] = (OT)((acc[mi][ni][r] + bv[ni]) * scale);
            }
}

// ---- Fused window attention: one block per (window, head), 4 waves ----------
__global__ __launch_bounds__(128) void attn_kernel(
    const _Float16* __restrict__ Q, const _Float16* __restrict__ KV,
    const float* __restrict__ bias_table, _Float16* __restrict__ X)
{
    const int bh = blockIdx.x;
    const int b  = bh / 12;
    const int h  = bh - b * 12;

    __shared__ _Float16 Qs[64 * 40];
    __shared__ _Float16 Ks[64 * 40];
    __shared__ _Float16 Vt[32 * 80];               // [d][j]
    __shared__ float    S [64 * 65];
    __shared__ _Float16 P [64 * 80];

    const int tid  = threadIdx.x;
    const int lane = tid & 31;
    const int wid  = tid >> 5;

    // Stage Q, K row-major [j][d]; V transposed [d][j]; zero-pad rows 49..63
    {
        const int r = tid >> 1;
        const int c = (tid & 1) * 16;
        if (r < 49) {
            const long base = (long)(b * 49 + r);
            const _Float16* q = Q  + base * 384 + h * 32 + c;
            const _Float16* k = KV + base * 768 + h * 32 + c;
            const _Float16* v = KV + base * 768 + 384 + h * 32 + c;
            *(v8h*)&Qs[r * 40 + c]     = *(const v8h*)(q);
            *(v8h*)&Qs[r * 40 + c + 8] = *(const v8h*)(q + 8);
            *(v8h*)&Ks[r * 40 + c]     = *(const v8h*)(k);
            *(v8h*)&Ks[r * 40 + c + 8] = *(const v8h*)(k + 8);
            v8h v0 = *(const v8h*)(v);
            v8h v1 = *(const v8h*)(v + 8);
            #pragma unroll
            for (int i = 0; i < 8; ++i) {
                Vt[(c + i) * 80 + r]     = v0[i];
                Vt[(c + 8 + i) * 80 + r] = v1[i];
            }
        } else {
            #pragma unroll
            for (int i = 0; i < 16; ++i) {
                Qs[r * 40 + c + i] = (_Float16)0;
                Ks[r * 40 + c + i] = (_Float16)0;
                Vt[(c + i) * 80 + r] = (_Float16)0;
            }
        }
    }
    __syncthreads();

    // S = Q * K^T
    {
        v16h aq = frag_a_lds(&Qs[wid * 16 * 40], 40, lane);
        const int mlane = (lane >> 4) << 3;
        const int nlane = lane & 15;
        #pragma unroll
        for (int nt = 0; nt < 4; ++nt) {
            v16h bk = frag_b_lds(&Ks[nt * 16 * 40], 40, lane);
            v8f s = {};
            s = wmma16(aq, bk, s);
            #pragma unroll
            for (int r = 0; r < 8; ++r)
                S[(wid * 16 + mlane + r) * 65 + nt * 16 + nlane] = s[r];
        }
    }
    __syncthreads();

    // Row softmax with analytic Swin relative-position bias
    if (tid < 64) {
        const int i = tid;
        if (i < 49) {
            const int yi = i / 7, xi = i - yi * 7;
            float mx = -1e30f;
            for (int j = 0; j < 49; ++j) {
                const int yj = j / 7, xj = j - yj * 7;
                const int rel = (yi - yj + 6) * 13 + (xi - xj + 6);
                const float s = S[i * 65 + j] + bias_table[rel * 12 + h];
                S[i * 65 + j] = s;
                mx = fmaxf(mx, s);
            }
            float sum = 0.f;
            for (int j = 0; j < 49; ++j) {
                const float e = __expf(S[i * 65 + j] - mx);
                S[i * 65 + j] = e;
                sum += e;
            }
            const float inv = 1.f / sum;
            for (int j = 0; j < 49; ++j) P[i * 80 + j] = (_Float16)(S[i * 65 + j] * inv);
            for (int j = 49; j < 64; ++j) P[i * 80 + j] = (_Float16)0;
        } else {
            for (int j = 0; j < 64; ++j) P[i * 80 + j] = (_Float16)0;
        }
    }
    __syncthreads();

    // O = P * V (K=64 in two steps, N=32 in two tiles)
    {
        v8f oacc[2] = {};
        #pragma unroll
        for (int ks = 0; ks < 2; ++ks) {
            v16h ap = frag_a_lds(&P[wid * 16 * 80 + ks * 32], 80, lane);
            #pragma unroll
            for (int nt = 0; nt < 2; ++nt) {
                v16h bv = frag_b_lds(&Vt[nt * 16 * 80 + ks * 32], 80, lane);
                oacc[nt] = wmma16(ap, bv, oacc[nt]);
            }
        }
        const int mlane = (lane >> 4) << 3;
        const int nlane = lane & 15;
        #pragma unroll
        for (int nt = 0; nt < 2; ++nt)
            #pragma unroll
            for (int r = 0; r < 8; ++r) {
                const int m = wid * 16 + mlane + r;
                if (m < 49)
                    X[((long)(b * 49 + m)) * 384 + h * 32 + nt * 16 + nlane] =
                        (_Float16)oacc[nt][r];
            }
    }
}

extern "C" void kernel_launch(void* const* d_in, const int* in_sizes, int n_in,
                              void* d_out, int out_size, void* d_ws, size_t ws_size,
                              hipStream_t stream) {
    (void)in_sizes; (void)n_in; (void)out_size; (void)ws_size;
    const float* x_edge = (const float*)d_in[0];
    const float* x_body = (const float*)d_in[1];
    const float* q_w    = (const float*)d_in[2];
    const float* q_b    = (const float*)d_in[3];
    const float* kv_w   = (const float*)d_in[4];
    const float* kv_b   = (const float*)d_in[5];
    const float* proj_w = (const float*)d_in[6];
    const float* proj_b = (const float*)d_in[7];
    const float* bias_t = (const float*)d_in[8];
    float* out = (float*)d_out;

    char* ws = (char*)d_ws;
    _Float16* Qb   = (_Float16*)(ws + WS_Q_OFF);
    _Float16* KVb  = (_Float16*)(ws + WS_KV_OFF);
    _Float16* Xb   = (_Float16*)(ws + WS_X_OFF);
    _Float16* WtQ  = (_Float16*)(ws + WS_WTQ_OFF);
    _Float16* WtKV = (_Float16*)(ws + WS_WTKV_OFF);
    _Float16* WtP  = (_Float16*)(ws + WS_WTP_OFF);

    const float scale = 0.17677669529663688f;      // 32^-0.5
    const int MB = 200704 / 64;                    // 3136 row blocks

    transpose_w_kernel<<<(384 * 384) / 256, 256, 0, stream>>>(q_w, WtQ, 384);
    transpose_w_kernel<<<(384 * 768) / 256, 256, 0, stream>>>(kv_w, WtKV, 768);
    transpose_w_kernel<<<(384 * 384) / 256, 256, 0, stream>>>(proj_w, WtP, 384);

    gemm_bias_kernel<float, _Float16><<<dim3(3, MB), 256, 0, stream>>>(
        x_edge, WtQ, q_b, Qb, 384, scale);
    gemm_bias_kernel<float, _Float16><<<dim3(6, MB), 256, 0, stream>>>(
        x_body, WtKV, kv_b, KVb, 768, 1.0f);
    attn_kernel<<<4096 * 12, 128, 0, stream>>>(Qb, KVb, bias_t, Xb);
    gemm_bias_kernel<_Float16, float><<<dim3(3, MB), 256, 0, stream>>>(
        Xb, WtP, proj_b, out, 384, 1.0f);
}
